// mGRU_41884521071209
// MI455X (gfx1250) — compile-verified, matrix operand order checked by reading, not run on hardware
//
#include <hip/hip_runtime.h>
#include <hip/hip_bf16.h>

// ---------------------------------------------------------------------------
// match-GRU (Wang-Jiang) forward for MI455X / gfx1250.
// All GEMMs: v_wmma_f32_16x16x32_bf16, operands pre-converted to bf16 so the
// inner loop is pure b128 loads + WMMA. Sequential scans = per-step launches.
// ---------------------------------------------------------------------------

constexpr int kTP = 64, kTH = 64, kB = 256, kIN = 768, kD = 512, kC = 3;

typedef __attribute__((ext_vector_type(16))) __bf16 bf16x16;
typedef __attribute__((ext_vector_type(8)))  __bf16 bf16x8;
typedef __attribute__((ext_vector_type(8)))  float  f32x8;

union Frag { bf16x16 v; bf16x8 h[2]; };

__device__ __forceinline__ float sigmoidf(float x) {
  return 1.0f / (1.0f + __expf(-x));
}

// ---------------------------------------------------------------------------
// WMMA GEMM, bf16 operands:  C[M,N] = act( A[M,K] * B^T + bias [+ C] )
//   A: (M,K) bf16 row-major, B: (N,K) bf16 row-major.
//   Cf: fp32 output (nullable). Cb: bf16 output (nullable).
// Wave tile: 16(M) x 32(N), two accumulators share one A fragment.
// Block: 8 waves -> 32(M) x 128(N). Requires M%32==0, N%128==0, K%32==0.
// Per-lane fragment = two contiguous 16B chunks (ISA 16-bit A/B layout).
// ---------------------------------------------------------------------------
__global__ __launch_bounds__(256) void wmma_gemm_bf16(
    const __bf16* __restrict__ A, int lda,
    const __bf16* __restrict__ B, int ldb,
    const float* __restrict__ bias,
    float* __restrict__ Cf, __bf16* __restrict__ Cb, int ldc,
    int M, int N, int K, int accum, int relu)
{
  const int lane = threadIdx.x & 31;
  const int wave = threadIdx.x >> 5;        // 0..7
  const int half = lane >> 4;               // K/M +-8 selector per ISA layout
  const int row  = lane & 15;

  const int m0 = blockIdx.y * 32  + (wave & 1) * 16;
  const int n0 = blockIdx.x * 128 + (wave >> 1) * 32;
  if (m0 >= M || n0 >= N) return;           // wave-uniform

  const __bf16* ar  = A + (size_t)(m0 + row)      * lda + half * 8;
  const __bf16* br0 = B + (size_t)(n0 + row)      * ldb + half * 8;
  const __bf16* br1 = B + (size_t)(n0 + 16 + row) * ldb + half * 8;

  f32x8 acc0 = {}, acc1 = {};

#pragma unroll 2
  for (int k0 = 0; k0 < K; k0 += 32) {
    Frag a, b0, b1;
    a.h[0]  = *(const bf16x8*)(ar  + k0);
    a.h[1]  = *(const bf16x8*)(ar  + k0 + 16);
    b0.h[0] = *(const bf16x8*)(br0 + k0);
    b0.h[1] = *(const bf16x8*)(br0 + k0 + 16);
    b1.h[0] = *(const bf16x8*)(br1 + k0);
    b1.h[1] = *(const bf16x8*)(br1 + k0 + 16);
    acc0 = __builtin_amdgcn_wmma_f32_16x16x32_bf16(
        false, a.v, false, b0.v, (short)0, acc0, false, false);
    acc1 = __builtin_amdgcn_wmma_f32_16x16x32_bf16(
        false, a.v, false, b1.v, (short)0, acc1, false, false);
  }

  // store: VGPR r -> M = m0 + 8*half + r ; N = n_tile + (lane&15)
#pragma unroll
  for (int tile = 0; tile < 2; ++tile) {
    const int n = n0 + tile * 16 + row;
    const float bv = bias ? bias[n] : 0.0f;
#pragma unroll
    for (int rI = 0; rI < 8; ++rI) {
      const int m = m0 + half * 8 + rI;
      const size_t idx = (size_t)m * ldc + n;
      float v = (tile ? acc1[rI] : acc0[rI]) + bv;
      if (accum && Cf) v += Cf[idx];
      if (relu)        v = fmaxf(v, 0.0f);
      if (Cf) Cf[idx] = v;
      if (Cb) Cb[idx] = (__bf16)v;
    }
  }
}

// ---------------------------------------------------------------------------
// Prep: fp32 -> bf16 flat convert / transposed convert / zero fill
// ---------------------------------------------------------------------------
__global__ void cvt_bf16_k(const float* __restrict__ in,
                           __bf16* __restrict__ out, int n) {
  int i = blockIdx.x * blockDim.x + threadIdx.x;
  if (i < n) out[i] = (__bf16)in[i];
}

// in: (Kd,Nd) row-major fp32  ->  out: (Nd,Kd) row-major bf16  (out = in^T)
__global__ void transpose_cvt_k(const float* __restrict__ in,
                                __bf16* __restrict__ out, int Kd, int Nd) {
  int i = blockIdx.x * blockDim.x + threadIdx.x;
  if (i >= Kd * Nd) return;
  const int k = i / Nd, n = i - k * Nd;
  out[(size_t)n * Kd + k] = (__bf16)in[i];
}

__global__ void zero_f_k(float* __restrict__ p, int n) {
  int i = blockIdx.x * blockDim.x + threadIdx.x;
  if (i < n) p[i] = 0.0f;
}
__global__ void zero_bf_k(__bf16* __restrict__ p, int n) {
  int i = blockIdx.x * blockDim.x + threadIdx.x;
  if (i < n) p[i] = (__bf16)0.0f;
}

// mask[t,b] = any(enc[t,b,:] != 0)
__global__ void compute_mask(const float* __restrict__ enc,
                             float* __restrict__ mask, int TB, int IN) {
  int idx = blockIdx.x * blockDim.x + threadIdx.x;
  if (idx >= TB) return;
  const float* p = enc + (size_t)idx * IN;
  float m = 0.0f;
  for (int i = 0; i < IN; ++i)
    if (p[i] != 0.0f) { m = 1.0f; break; }
  mask[idx] = m;
}

// ---------------------------------------------------------------------------
// Masked GRU step (torch cell). Gi/Gh: (B,3D) gates [r,z,n].
// Updates h (fp32 + bf16 shadow), writes o_t (fp32 + bf16 shadow).
// ---------------------------------------------------------------------------
__global__ void gru_step_update(const float* __restrict__ Gi,
                                const float* __restrict__ Gh,
                                const float* __restrict__ mask_t,
                                float* __restrict__ h, __bf16* __restrict__ h_bf,
                                float* __restrict__ o_t, __bf16* __restrict__ o_t_bf,
                                const float* __restrict__ o_prev, int first)
{
  int idx = blockIdx.x * blockDim.x + threadIdx.x;
  if (idx >= kB * kD) return;
  const int b = idx / kD, d = idx - b * kD;
  const size_t g = (size_t)b * (3 * kD);
  const float ir = Gi[g + d], iz = Gi[g + kD + d], ig = Gi[g + 2 * kD + d];
  const float hr = Gh[g + d], hz = Gh[g + kD + d], hg = Gh[g + 2 * kD + d];
  const float rr = sigmoidf(ir + hr);
  const float zz = sigmoidf(iz + hz);
  const float nn = tanhf(ig + rr * hg);
  const float hv = h[idx];
  const float raw = (1.0f - zz) * nn + zz * hv;
  const float m = mask_t[b];
  const float hnew = m * raw + (1.0f - m) * hv;
  h[idx] = hnew;  h_bf[idx] = (__bf16)hnew;
  const float onew = first ? raw : (m * raw + (1.0f - m) * o_prev[idx]);
  o_t[idx] = onew;  o_t_bf[idx] = (__bf16)onew;
}

// Match-GRU step: carries only r (fp32 + bf16 shadow).
__global__ void mgru_update(const float* __restrict__ Gi,
                            const float* __restrict__ Gh,
                            const float* __restrict__ mask_t,
                            float* __restrict__ r, __bf16* __restrict__ r_bf)
{
  int idx = blockIdx.x * blockDim.x + threadIdx.x;
  if (idx >= kB * kD) return;
  const int b = idx / kD, d = idx - b * kD;
  const size_t g = (size_t)b * (3 * kD);
  const float ir = Gi[g + d], iz = Gi[g + kD + d], ig = Gi[g + 2 * kD + d];
  const float hr = Gh[g + d], hz = Gh[g + kD + d], hg = Gh[g + 2 * kD + d];
  const float rr = sigmoidf(ir + hr);
  const float zz = sigmoidf(iz + hz);
  const float nn = tanhf(ig + rr * hg);
  const float rv = r[idx];
  const float raw = (1.0f - zz) * nn + zz * rv;
  const float m = mask_t[b];
  const float rnew = m * raw + (1.0f - m) * rv;
  r[idx] = rnew;  r_bf[idx] = (__bf16)rnew;
}

// ---------------------------------------------------------------------------
// Attention step: one workgroup per batch element b.
//   score[p] = sum_d tanh(Whs[p,b,d] + s[b,d]) * w_e[d]  - 1000*(1-mask_p)
//   alpha = softmax_p(score);  a_t[b,d] = sum_p alpha[p] * o_p[p,b,d]
//   mt_bf[b,0:D] = bf16(a_t) ;  mt_bf[b,D:2D] = bf16(h_t)
// ---------------------------------------------------------------------------
__global__ __launch_bounds__(256) void attention_step(
    const float* __restrict__ Whs,    // (TP,B,D) fp32
    const float* __restrict__ s,      // (B,D)    fp32
    const float* __restrict__ w_e,    // (D)
    const float* __restrict__ o_p,    // (TP,B,D) fp32
    const float* __restrict__ mask_p, // (TP,B)
    const float* __restrict__ h_t,    // (B,D)    fp32
    __bf16* __restrict__ mt_bf)       // (B,2D)   bf16
{
  __shared__ float s_sh[kD];
  __shared__ float we_sh[kD];
  __shared__ float score[kTP];
  const int b = blockIdx.x;
  const int tid = threadIdx.x;

  for (int d = tid; d < kD; d += 256) {
    s_sh[d]  = s[(size_t)b * kD + d];
    we_sh[d] = w_e[d];
  }
  __syncthreads();

  const int lane = tid & 31, wave = tid >> 5;
  for (int p = wave; p < kTP; p += 8) {
    const float* wrow = Whs + ((size_t)p * kB + b) * kD;
    float loc = 0.0f;
    for (int d = lane; d < kD; d += 32)
      loc += tanhf(wrow[d] + s_sh[d]) * we_sh[d];
#pragma unroll
    for (int off = 16; off; off >>= 1) loc += __shfl_xor(loc, off, 32);
    if (lane == 0)
      score[p] = loc - 1000.0f * (1.0f - mask_p[(size_t)p * kB + b]);
  }
  __syncthreads();

  if (tid == 0) {                  // softmax over 64 entries (serial, cheap)
    float mx = -3.0e38f;
    for (int p = 0; p < kTP; ++p) mx = fmaxf(mx, score[p]);
    float den = 0.0f;
    for (int p = 0; p < kTP; ++p) { float e = __expf(score[p] - mx); score[p] = e; den += e; }
    const float inv = 1.0f / den;
    for (int p = 0; p < kTP; ++p) score[p] *= inv;
  }
  __syncthreads();

  for (int d = tid; d < kD; d += 256) {
    float acc = 0.0f;
    for (int p = 0; p < kTP; ++p)
      acc += score[p] * o_p[((size_t)p * kB + b) * kD + d];
    mt_bf[(size_t)b * (2 * kD) + d]      = (__bf16)acc;
    mt_bf[(size_t)b * (2 * kD) + kD + d] = (__bf16)h_t[(size_t)b * kD + d];
  }
}

// out[b,c] = relu( r[b,:] . W_out[c,:] + b_out[c] ), c < 3
__global__ void final_out_k(const float* __restrict__ r,
                            const float* __restrict__ W_out,
                            const float* __restrict__ b_out,
                            float* __restrict__ out)
{
  int idx = blockIdx.x * blockDim.x + threadIdx.x;
  if (idx >= kB * kC) return;
  const int b = idx / kC, c = idx - b * kC;
  float acc = b_out[c];
  const float* rr = r + (size_t)b * kD;
  const float* wr = W_out + (size_t)c * kD;
  for (int d = 0; d < kD; ++d) acc += rr[d] * wr[d];
  out[idx] = fmaxf(acc, 0.0f);
}

// ---------------------------------------------------------------------------
// Host orchestration
// ---------------------------------------------------------------------------
static inline dim3 gemm_grid(int M, int N) {
  return dim3((unsigned)(N / 128), (unsigned)(M / 32), 1);
}
static inline unsigned blocks(int n) { return (unsigned)((n + 255) / 256); }

extern "C" void kernel_launch(void* const* d_in, const int* in_sizes, int n_in,
                              void* d_out, int out_size, void* d_ws, size_t ws_size,
                              hipStream_t stream) {
  const float* enc_p = (const float*)d_in[0];
  const float* enc_h = (const float*)d_in[1];
  const float* W_c   = (const float*)d_in[2];
  const float* b_c   = (const float*)d_in[3];
  const float* Wp_ih = (const float*)d_in[4];
  const float* Wp_hh = (const float*)d_in[5];
  const float* bp_ih = (const float*)d_in[6];
  const float* bp_hh = (const float*)d_in[7];
  const float* Wh_ih = (const float*)d_in[8];
  const float* Wh_hh = (const float*)d_in[9];
  const float* bh_ih = (const float*)d_in[10];
  const float* bh_hh = (const float*)d_in[11];
  const float* W_s   = (const float*)d_in[12];
  const float* W_t   = (const float*)d_in[13];
  const float* W_m   = (const float*)d_in[14];
  const float* w_e   = (const float*)d_in[15];
  const float* Wm_ih = (const float*)d_in[16];
  const float* Wm_hh = (const float*)d_in[17];
  const float* bm_ih = (const float*)d_in[18];
  const float* bm_hh = (const float*)d_in[19];
  const float* W_out = (const float*)d_in[20];
  const float* b_out = (const float*)d_in[21];
  (void)in_sizes; (void)n_in; (void)out_size; (void)ws_size;

  // ---- workspace layout: fp32 region, then bf16 region -------------------
  // (every buffer size is a multiple of 16 elements -> 16B alignment holds)
  float* wf = (float*)d_ws;
  float* o_p    = wf; wf += (size_t)kTP * kB * kD;   // premise GRU outputs
  float* o_h    = wf; wf += (size_t)kTH * kB * kD;   // hypothesis GRU outputs
  float* Whs    = wf; wf += (size_t)kTP * kB * kD;   // Hs @ W_s
  float* h_p    = wf; wf += (size_t)kB * kD;
  float* h_h    = wf; wf += (size_t)kB * kD;
  float* r_st   = wf; wf += (size_t)kB * kD;
  float* s_buf  = wf; wf += (size_t)kB * kD;
  float* gi_t   = wf; wf += (size_t)kB * 3 * kD;
  float* gh_t   = wf; wf += (size_t)kB * 3 * kD;
  float* mask_p = wf; wf += (size_t)kTP * kB;
  float* mask_h = wf; wf += (size_t)kTH * kB;

  __bf16* wb = (__bf16*)wf;
  __bf16* encp_bf  = wb; wb += (size_t)kTP * kB * kIN;
  __bf16* ench_bf  = wb; wb += (size_t)kTH * kB * kIN;
  __bf16* ep_bf    = wb; wb += (size_t)kTP * kB * kD;
  __bf16* eh_bf    = wb; wb += (size_t)kTH * kB * kD;
  __bf16* op_bf    = wb; wb += (size_t)kTP * kB * kD;
  __bf16* oh_bf    = wb; wb += (size_t)kTH * kB * kD;
  __bf16* hp_bf    = wb; wb += (size_t)kB * kD;
  __bf16* hh_bf    = wb; wb += (size_t)kB * kD;
  __bf16* r_bf     = wb; wb += (size_t)kB * kD;
  __bf16* mt_bf    = wb; wb += (size_t)kB * 2 * kD;
  __bf16* Wc_bf    = wb; wb += (size_t)kD * kIN;
  __bf16* Wpih_bf  = wb; wb += (size_t)3 * kD * kD;
  __bf16* Wphh_bf  = wb; wb += (size_t)3 * kD * kD;
  __bf16* Whih_bf  = wb; wb += (size_t)3 * kD * kD;
  __bf16* Whhh_bf  = wb; wb += (size_t)3 * kD * kD;
  __bf16* Wmih_bf  = wb; wb += (size_t)3 * kD * 2 * kD;
  __bf16* Wmhh_bf  = wb; wb += (size_t)3 * kD * kD;
  __bf16* WsT_bf   = wb; wb += (size_t)kD * kD;      // (N,K) = W_s^T
  __bf16* WtT_bf   = wb; wb += (size_t)kD * kD;
  __bf16* WmT_bf   = wb; wb += (size_t)kD * kD;

  const dim3 blk(256);

  // ---- prep: bf16 conversions / transposes -------------------------------
  cvt_bf16_k<<<blocks(kTP * kB * kIN), blk, 0, stream>>>(enc_p, encp_bf, kTP * kB * kIN);
  cvt_bf16_k<<<blocks(kTH * kB * kIN), blk, 0, stream>>>(enc_h, ench_bf, kTH * kB * kIN);
  cvt_bf16_k<<<blocks(kD * kIN), blk, 0, stream>>>(W_c, Wc_bf, kD * kIN);
  cvt_bf16_k<<<blocks(3 * kD * kD), blk, 0, stream>>>(Wp_ih, Wpih_bf, 3 * kD * kD);
  cvt_bf16_k<<<blocks(3 * kD * kD), blk, 0, stream>>>(Wp_hh, Wphh_bf, 3 * kD * kD);
  cvt_bf16_k<<<blocks(3 * kD * kD), blk, 0, stream>>>(Wh_ih, Whih_bf, 3 * kD * kD);
  cvt_bf16_k<<<blocks(3 * kD * kD), blk, 0, stream>>>(Wh_hh, Whhh_bf, 3 * kD * kD);
  cvt_bf16_k<<<blocks(3 * kD * 2 * kD), blk, 0, stream>>>(Wm_ih, Wmih_bf, 3 * kD * 2 * kD);
  cvt_bf16_k<<<blocks(3 * kD * kD), blk, 0, stream>>>(Wm_hh, Wmhh_bf, 3 * kD * kD);
  transpose_cvt_k<<<blocks(kD * kD), blk, 0, stream>>>(W_s, WsT_bf, kD, kD);
  transpose_cvt_k<<<blocks(kD * kD), blk, 0, stream>>>(W_t, WtT_bf, kD, kD);
  transpose_cvt_k<<<blocks(kD * kD), blk, 0, stream>>>(W_m, WmT_bf, kD, kD);

  // ---- masks -------------------------------------------------------------
  compute_mask<<<blocks(kTP * kB), blk, 0, stream>>>(enc_p, mask_p, kTP * kB, kIN);
  compute_mask<<<blocks(kTH * kB), blk, 0, stream>>>(enc_h, mask_h, kTH * kB, kIN);

  // ---- compress: e = enc @ W_c^T + b_c  (bf16 output only) ---------------
  wmma_gemm_bf16<<<gemm_grid(kTP * kB, kD), blk, 0, stream>>>(
      encp_bf, kIN, Wc_bf, kIN, b_c, nullptr, ep_bf, kD, kTP * kB, kD, kIN, 0, 0);
  wmma_gemm_bf16<<<gemm_grid(kTH * kB, kD), blk, 0, stream>>>(
      ench_bf, kIN, Wc_bf, kIN, b_c, nullptr, eh_bf, kD, kTH * kB, kD, kIN, 0, 0);

  // ---- zero recurrent states ---------------------------------------------
  zero_f_k<<<blocks(kB * kD), blk, 0, stream>>>(h_p, kB * kD);
  zero_f_k<<<blocks(kB * kD), blk, 0, stream>>>(h_h, kB * kD);
  zero_f_k<<<blocks(kB * kD), blk, 0, stream>>>(r_st, kB * kD);
  zero_bf_k<<<blocks(kB * kD), blk, 0, stream>>>(hp_bf, kB * kD);
  zero_bf_k<<<blocks(kB * kD), blk, 0, stream>>>(hh_bf, kB * kD);
  zero_bf_k<<<blocks(kB * kD), blk, 0, stream>>>(r_bf, kB * kD);

  // ---- premise GRU over TP steps -----------------------------------------
  for (int t = 0; t < kTP; ++t) {
    wmma_gemm_bf16<<<gemm_grid(kB, 3 * kD), blk, 0, stream>>>(
        ep_bf + (size_t)t * kB * kD, kD, Wpih_bf, kD, bp_ih,
        gi_t, nullptr, 3 * kD, kB, 3 * kD, kD, 0, 0);
    wmma_gemm_bf16<<<gemm_grid(kB, 3 * kD), blk, 0, stream>>>(
        hp_bf, kD, Wphh_bf, kD, bp_hh,
        gh_t, nullptr, 3 * kD, kB, 3 * kD, kD, 0, 0);
    gru_step_update<<<blocks(kB * kD), blk, 0, stream>>>(
        gi_t, gh_t, mask_p + (size_t)t * kB, h_p, hp_bf,
        o_p + (size_t)t * kB * kD, op_bf + (size_t)t * kB * kD,
        t ? (o_p + (size_t)(t - 1) * kB * kD) : o_p, t == 0);
  }

  // ---- hypothesis GRU over TH steps --------------------------------------
  for (int t = 0; t < kTH; ++t) {
    wmma_gemm_bf16<<<gemm_grid(kB, 3 * kD), blk, 0, stream>>>(
        eh_bf + (size_t)t * kB * kD, kD, Whih_bf, kD, bh_ih,
        gi_t, nullptr, 3 * kD, kB, 3 * kD, kD, 0, 0);
    wmma_gemm_bf16<<<gemm_grid(kB, 3 * kD), blk, 0, stream>>>(
        hh_bf, kD, Whhh_bf, kD, bh_hh,
        gh_t, nullptr, 3 * kD, kB, 3 * kD, kD, 0, 0);
    gru_step_update<<<blocks(kB * kD), blk, 0, stream>>>(
        gi_t, gh_t, mask_h + (size_t)t * kB, h_h, hh_bf,
        o_h + (size_t)t * kB * kD, oh_bf + (size_t)t * kB * kD,
        t ? (o_h + (size_t)(t - 1) * kB * kD) : o_h, t == 0);
  }

  // ---- Whs = Hs @ W_s  (fp32 output, reused 64x) -------------------------
  wmma_gemm_bf16<<<gemm_grid(kTP * kB, kD), blk, 0, stream>>>(
      op_bf, kD, WsT_bf, kD, nullptr, Whs, nullptr, kD, kTP * kB, kD, kD, 0, 0);

  // ---- match-GRU with attention over TH steps ----------------------------
  for (int t = 0; t < kTH; ++t) {
    const float*  h_t    = o_h  + (size_t)t * kB * kD;
    const __bf16* h_t_bf = oh_bf + (size_t)t * kB * kD;
    // s = h_t @ W_t + r @ W_m
    wmma_gemm_bf16<<<gemm_grid(kB, kD), blk, 0, stream>>>(
        h_t_bf, kD, WtT_bf, kD, nullptr, s_buf, nullptr, kD, kB, kD, kD, 0, 0);
    wmma_gemm_bf16<<<gemm_grid(kB, kD), blk, 0, stream>>>(
        r_bf, kD, WmT_bf, kD, nullptr, s_buf, nullptr, kD, kB, kD, kD, 1, 0);
    // softmax attention + build mt = [a_t, h_t] (bf16)
    attention_step<<<kB, blk, 0, stream>>>(Whs, s_buf, w_e, o_p, mask_p, h_t, mt_bf);
    // GRU cell: gi = mt @ Wm_ih^T + bm_ih ; gh = r @ Wm_hh^T + bm_hh
    wmma_gemm_bf16<<<gemm_grid(kB, 3 * kD), blk, 0, stream>>>(
        mt_bf, 2 * kD, Wmih_bf, 2 * kD, bm_ih,
        gi_t, nullptr, 3 * kD, kB, 3 * kD, 2 * kD, 0, 0);
    wmma_gemm_bf16<<<gemm_grid(kB, 3 * kD), blk, 0, stream>>>(
        r_bf, kD, Wmhh_bf, kD, bm_hh,
        gh_t, nullptr, 3 * kD, kB, 3 * kD, kD, 0, 0);
    mgru_update<<<blocks(kB * kD), blk, 0, stream>>>(
        gi_t, gh_t, mask_h + (size_t)t * kB, r_st, r_bf);
  }

  // ---- head: relu(r @ W_out^T + b_out) -----------------------------------
  final_out_k<<<blocks(kB * kC), blk, 0, stream>>>(
      r_st, W_out, b_out, (float*)d_out);
}
